// Model_39676907884345
// MI455X (gfx1250) — matrix operand order repaired and checked by reading, not compiled, not run on hardware
//
#include <hip/hip_runtime.h>

// ---------------------------------------------------------------------------
// Flash attention for MI455X (gfx1250, wave32).
//   S = (Q K^T) / temperature ; P = softmax(S) ; O = P V
// Compute-bound (~512 FLOP/byte) -> WMMA f16 inputs / f32 accumulation
// (v_wmma_f32_16x16x32_f16). Block = 128 threads (4 waves); each wave owns
// 16 query rows. K staged f32->f16 row-major in LDS (conflict-free b64
// stores); V staged TRANSPOSED via per-lane column-strip loads (coalesced
// b32 gathers, 2-way-max h4 stores with 72-half padded stride; fragment
// reads tile all 64 banks). 1/temp * log2(e) folded into Q before f16
// conversion; softmax runs on exp2 (bit-equivalent after normalization).
// ---------------------------------------------------------------------------

typedef __attribute__((ext_vector_type(16))) _Float16 v16h;
typedef __attribute__((ext_vector_type(8)))  _Float16 h8;
typedef __attribute__((ext_vector_type(4)))  _Float16 h4;
typedef __attribute__((ext_vector_type(8)))  float    v8f;
typedef __attribute__((ext_vector_type(4)))  float    f4;

static constexpr int kB   = 2;
static constexpr int kH   = 16;
static constexpr int kSQ  = 2048;
static constexpr int kSKV = 2048;
static constexpr int kD   = 128;
static constexpr int BM   = 64;   // query rows per block (4 waves x 16)
static constexpr int BN   = 64;   // kv rows per iteration
static constexpr int NW   = 4;    // waves per block
static constexpr int VTS  = 72;   // Vt row stride in halves: 144B, 16B-aligned,
                                  // 36 dwords -> 2-way store / 0-way load banks

__global__ __launch_bounds__(128)
void fattn_kernel(const float* __restrict__ qg, const float* __restrict__ kg,
                  const float* __restrict__ vg, const float* __restrict__ tg,
                  float* __restrict__ outg)
{
    __shared__ alignas(32) _Float16 Ksh[BN][kD];      // 16 KB
    __shared__ alignas(32) _Float16 Vt[kD][VTS];      // 18 KB (transposed V)
    __shared__ alignas(32) _Float16 Psh[NW][16][BN];  //  8 KB (per-wave P)

    const int t    = threadIdx.x;
    const int lane = t & 31;
    const int w    = t >> 5;
    const int half = lane >> 4;   // 0: lanes 0-15, 1: lanes 16-31
    const int l15  = lane & 15;

    const int bh = blockIdx.y;            // b*H + h
    const int h  = bh & (kH - 1);
    const int q0 = blockIdx.x * BM + w * 16;

    // fold 1/temperature and log2(e) into Q (exact f32 mul before f16 cvt);
    // softmax then uses exp2 -> identical result after normalization.
    const float qscale = (1.0f / tg[h]) * 1.44269504088896340736f;

    const float* qbase = qg + (long)bh * kSQ  * kD;
    const float* kbase = kg + (long)bh * kSKV * kD;
    const float* vbase = vg + (long)bh * kSKV * kD;

    // ---- Q fragments: A operand of v_wmma_f32_16x16x32_f16, this code's contiguous k order -----------------
    // lane L: M = L&15; halves 0-15 = K base..base+15  (base = 16*(L>=16))
    //                   per K-chunk c.
    v16h qa[4];
    {
        const float* qr = qbase + (long)(q0 + l15) * kD;
#pragma unroll
        for (int c = 0; c < 4; ++c) {
            const int cb = 32 * c + half * 16;
            f4 x0 = *(const f4*)(qr + cb);
            f4 x1 = *(const f4*)(qr + cb + 4);
            f4 x2 = *(const f4*)(qr + cb + 8);
            f4 x3 = *(const f4*)(qr + cb + 12);
#pragma unroll
            for (int e = 0; e < 4; ++e) {
                qa[c][e]      = (_Float16)(x0[e] * qscale);
                qa[c][4 + e]  = (_Float16)(x1[e] * qscale);
                qa[c][8 + e]  = (_Float16)(x2[e] * qscale);
                qa[c][12 + e] = (_Float16)(x3[e] * qscale);
            }
        }
    }

    v8f o[8];                       // O accum: 16 rows x 128 cols (8 C-frags)
    float mrow[8], lrow[8];
#pragma unroll
    for (int df = 0; df < 8; ++df)
#pragma unroll
        for (int r = 0; r < 8; ++r) o[df][r] = 0.0f;
#pragma unroll
    for (int r = 0; r < 8; ++r) { mrow[r] = -3.0e30f; lrow[r] = 0.0f; }

    for (int j = 0; j < kSKV / BN; ++j) {
        const int kv0 = j * BN;

        __syncthreads();
        // ---- stage K row-major: coalesced b128 loads, conflict-free b64 st --
#pragma unroll
        for (int i = 0; i < 16; ++i) {
            const int idx = t + 128 * i;        // 0..2047 float4s
            const int row = idx >> 5;           // kv row within tile
            const int c4  = (idx & 31) * 4;     // d column (multiple of 4)
            f4 kk = *(const f4*)(kbase + (long)(kv0 + row) * kD + c4);
            h4 khh;
#pragma unroll
            for (int e = 0; e < 4; ++e) khh[e] = (_Float16)kk[e];
            *(h4*)&Ksh[row][c4] = khh;
        }
        // ---- stage V transposed: per-lane column strips ---------------------
        // lane holds V[rowb..rowb+3][dcol]; each k-row load is a contiguous
        // 128B line across the wave; single h4 store per strip (2-way max).
#pragma unroll
        for (int it = 0; it < 16; ++it) {
            const int combo = w + 4 * it;          // 0..63
            const int rowb  = (combo & 15) * 4;    // kv row base (x4)
            const int dcol  = (combo >> 4) * 32 + lane;
            const float* vp = vbase + (long)(kv0 + rowb) * kD + dcol;
            h4 vh;
#pragma unroll
            for (int kk2 = 0; kk2 < 4; ++kk2)
                vh[kk2] = (_Float16)vp[(long)kk2 * kD];
            *(h4*)&Vt[dcol][rowb] = vh;
        }
        // prefetch next KV tile into cache while we compute this one
        if (j + 1 < kSKV / BN) {
            const float* kn = kbase + (long)(kv0 + BN) * kD;
            const float* vn = vbase + (long)(kv0 + BN) * kD;
            __builtin_prefetch(kn + t * 64, 0, 1);   // 128 thr * 256B = 32KB
            __builtin_prefetch(vn + t * 64, 0, 1);
        }
        __syncthreads();

        // ---- S = Q K^T (16x64), 4 N-frags x 4 K-chunks = 16 WMMAs ----------
        v8f s[4];
#pragma unroll
        for (int nf = 0; nf < 4; ++nf) {
            v8f acc;
#pragma unroll
            for (int r = 0; r < 8; ++r) acc[r] = 0.0f;
#pragma unroll
            for (int c = 0; c < 4; ++c) {
                const int kb = 32 * c + half * 16;   // B: K base per lane half
                v16h bf = *(const v16h*)&Ksh[nf * 16 + l15][kb];
                acc = __builtin_amdgcn_wmma_f32_16x16x32_f16(
                        false, qa[c], false, bf, (short)0, acc, false, false);
            }
            s[nf] = acc;          // already scaled: logits in log2 units
        }

        // ---- online softmax (C-layout: VGPR r, lanes n<16 -> row r,
        //      lanes >=16 -> row r+8; row spans 16 lanes of one half) --------
        float mnew[8], corr[8];
#pragma unroll
        for (int r = 0; r < 8; ++r) {
            float mx = fmaxf(fmaxf(s[0][r], s[1][r]), fmaxf(s[2][r], s[3][r]));
#pragma unroll
            for (int msk = 8; msk >= 1; msk >>= 1)
                mx = fmaxf(mx, __shfl_xor(mx, msk));     // stays within half
            mnew[r] = fmaxf(mrow[r], mx);
            corr[r] = exp2f(mrow[r] - mnew[r]);
            mrow[r] = mnew[r];
        }
#pragma unroll
        for (int r = 0; r < 8; ++r) {
            const int prow = r + 8 * half;
            float psum = 0.0f;
#pragma unroll
            for (int nf = 0; nf < 4; ++nf) {
                float p = exp2f(s[nf][r] - mnew[r]);
                psum += p;
                Psh[w][prow][nf * 16 + l15] = (_Float16)p;
            }
#pragma unroll
            for (int msk = 8; msk >= 1; msk >>= 1)
                psum += __shfl_xor(psum, msk);
            lrow[r] = lrow[r] * corr[r] + psum;
        }
#pragma unroll
        for (int df = 0; df < 8; ++df)
#pragma unroll
            for (int r = 0; r < 8; ++r) o[df][r] *= corr[r];

        // cross-lane LDS RAW within the wave (P store -> P fragment load)
        asm volatile("s_wait_dscnt 0x0" ::: "memory");

        // ---- O += P V : 2 K-chunks x 8 D-frags = 16 WMMAs ------------------
#pragma unroll
        for (int c2 = 0; c2 < 2; ++c2) {
            const int pb = 32 * c2 + half * 16;      // A: K base per lane half
            h8 p0 = *(const h8*)&Psh[w][l15][pb];
            h8 p1 = *(const h8*)&Psh[w][l15][pb + 8];
            v16h af;
#pragma unroll
            for (int e = 0; e < 8; ++e) { af[e] = p0[e]; af[8 + e] = p1[e]; }
            const int kb2 = 32 * c2 + half * 16;     // B: K base per lane half
#pragma unroll
            for (int df = 0; df < 8; ++df) {
                const _Float16* vr = &Vt[df * 16 + l15][kb2];
                h8 b0 = *(const h8*)vr;              // 16B-aligned (stride 144B)
                h8 b1 = *(const h8*)(vr + 8);
                v16h bf;
#pragma unroll
                for (int e = 0; e < 8; ++e) { bf[e] = b0[e]; bf[8 + e] = b1[e]; }
                o[df] = __builtin_amdgcn_wmma_f32_16x16x32_f16(
                        false, af, false, bf, (short)0, o[df], false, false);
            }
        }
    }

    // ---- epilogue: O / l -> fp32 output ------------------------------------
#pragma unroll
    for (int r = 0; r < 8; ++r) {
        const int gq = q0 + r + 8 * half;
        const float inv_l = 1.0f / lrow[r];
        float* orow = outg + ((long)bh * kSQ + gq) * kD;
#pragma unroll
        for (int df = 0; df < 8; ++df)
            orow[df * 16 + l15] = o[df][r] * inv_l;
    }
}

extern "C" void kernel_launch(void* const* d_in, const int* in_sizes, int n_in,
                              void* d_out, int out_size, void* d_ws, size_t ws_size,
                              hipStream_t stream) {
    (void)in_sizes; (void)n_in; (void)out_size; (void)d_ws; (void)ws_size;
    const float* q = (const float*)d_in[0];
    const float* k = (const float*)d_in[1];
    const float* v = (const float*)d_in[2];
    const float* temp = (const float*)d_in[3];
    float* out = (float*)d_out;

    dim3 grid(kSQ / BM, kB * kH);
    dim3 block(128);
    hipLaunchKernelGGL(fattn_kernel, grid, block, 0, stream, q, k, v, temp, out);
}
